// EncP_53549652246657
// MI455X (gfx1250) — compile-verified, matrix operand order checked by reading, not run on hardware
//
#include <hip/hip_runtime.h>
#include <cstdint>
#include <cstddef>

// ---------------------------------------------------------------------------
// Model constants (from reference)
// ---------------------------------------------------------------------------
#define BATCH 16
#define NPTS  2048
#define KNN_K 24

// ---------------------------------------------------------------------------
// bf16 helpers (raw uint16 storage, RNE convert)
// ---------------------------------------------------------------------------
__device__ __forceinline__ float bf2f(uint16_t h) {
    unsigned int u = ((unsigned int)h) << 16;
    float f;
    __builtin_memcpy(&f, &u, 4);
    return f;
}
__device__ __forceinline__ uint16_t f2bf(float f) {
    unsigned int u;
    __builtin_memcpy(&u, &f, 4);
    unsigned int r = u + 0x7FFFu + ((u >> 16) & 1u);
    return (uint16_t)(r >> 16);
}

// ---------------------------------------------------------------------------
// WMMA fragment types (CDNA5, wave32)
// ---------------------------------------------------------------------------
typedef __attribute__((ext_vector_type(16))) __bf16 v16bf;
typedef __attribute__((ext_vector_type(8)))  float  v8f;
typedef __attribute__((ext_vector_type(4)))  unsigned int v4u;
typedef __attribute__((ext_vector_type(8)))  int v8i;
typedef __attribute__((ext_vector_type(4)))  int v4i;

union Frag16 {
    v16bf v;
    unsigned long long d[4];
};

#if __has_builtin(__builtin_amdgcn_tensor_load_to_lds)
#define HAVE_TDM 1
#else
#define HAVE_TDM 0
#endif

// ---------------------------------------------------------------------------
// 1) float -> bf16 conversion
// ---------------------------------------------------------------------------
__global__ void k_cvt_bf16(const float* __restrict__ src, uint16_t* __restrict__ dst, int n) {
    int i = blockIdx.x * blockDim.x + threadIdx.x;
    if (i < n) dst[i] = f2bf(src[i]);
}

// ---------------------------------------------------------------------------
// 2) fold BN (+conv bias) into per-channel scale/shift
// ---------------------------------------------------------------------------
__global__ void k_prep_ss(const float* __restrict__ g, const float* __restrict__ b,
                          const float* __restrict__ m, const float* __restrict__ v,
                          const float* __restrict__ bias,
                          float* __restrict__ s, float* __restrict__ t, int C) {
    int i = blockIdx.x * blockDim.x + threadIdx.x;
    if (i >= C) return;
    float sc = g[i] * rsqrtf(v[i] + 1e-5f);
    float bi = bias ? bias[i] : 0.0f;
    s[i] = sc;
    t[i] = (bi - m[i]) * sc + b[i];
}

// ---------------------------------------------------------------------------
// 3) embed: x(B,3,N) -> act(B*N, 72) bf16 token-major, relu(bn(w.x))
// ---------------------------------------------------------------------------
__global__ void k_embed(const float* __restrict__ x, const float* __restrict__ w,
                        const float* __restrict__ s, const float* __restrict__ t,
                        uint16_t* __restrict__ act, int N) {
    int idx = blockIdx.x * blockDim.x + threadIdx.x;
    int total = BATCH * N * 72;
    if (idx >= total) return;
    int o  = idx % 72;
    int bn = idx / 72;
    int n  = bn % N;
    int b  = bn / N;
    const float* xb = x + (size_t)b * 3 * N;
    float acc = w[o * 3 + 0] * xb[n] + w[o * 3 + 1] * xb[N + n] + w[o * 3 + 2] * xb[2 * N + n];
    float vv = acc * s[o] + t[o];
    act[(size_t)bn * 72 + o] = f2bf(vv > 0.f ? vv : 0.f);
}

// ---------------------------------------------------------------------------
// 4) furthest-point sampling: one workgroup per batch, sequential over G
// ---------------------------------------------------------------------------
__global__ void k_fps(const float* __restrict__ xyz, int N, int G,
                      int* __restrict__ fidx, float* __restrict__ lc) {
    __shared__ float dmin[NPTS];
    __shared__ float rv[256];
    __shared__ int   ri[256];
    __shared__ int   sfar;
    int b = blockIdx.x, tid = threadIdx.x;
    const float* pts = xyz + (size_t)b * N * 3;
    for (int n = tid; n < N; n += 256) dmin[n] = 1e10f;
    if (tid == 0) sfar = 0;
    __syncthreads();
    for (int j = 0; j < G; j++) {
        int far = sfar;
        float cx = pts[far * 3], cy = pts[far * 3 + 1], cz = pts[far * 3 + 2];
        if (tid == 0) {
            fidx[b * G + j] = far;
            float* o = lc + ((size_t)b * G + j) * 3;
            o[0] = cx; o[1] = cy; o[2] = cz;
        }
        float bv = -1.0f; int bi = 0;
        for (int n = tid; n < N; n += 256) {
            float dx = pts[n * 3] - cx, dy = pts[n * 3 + 1] - cy, dz = pts[n * 3 + 2] - cz;
            float nd = dx * dx + dy * dy + dz * dz;
            float dm = dmin[n];
            dm = nd < dm ? nd : dm;
            dmin[n] = dm;
            if (dm > bv) { bv = dm; bi = n; }
        }
        rv[tid] = bv; ri[tid] = bi;
        __syncthreads();
        for (int st = 128; st > 0; st >>= 1) {
            if (tid < st) {
                float ov = rv[tid + st]; int oi = ri[tid + st];
                if (ov > rv[tid] || (ov == rv[tid] && oi < ri[tid])) { rv[tid] = ov; ri[tid] = oi; }
            }
            __syncthreads();
        }
        if (tid == 0) sfar = ri[0];
        __syncthreads();
    }
}

// ---------------------------------------------------------------------------
// 5) kNN: one workgroup per (b,g) query; K sequential arg-mins over LDS dists
// ---------------------------------------------------------------------------
__global__ void k_knn(const float* __restrict__ xyz, const float* __restrict__ q,
                      int N, int G, int* __restrict__ kidx) {
    __shared__ float dist[NPTS];
    __shared__ float rv[256];
    __shared__ int   ri[256];
    int bg = blockIdx.x, tid = threadIdx.x;
    int b = bg / G;
    const float* pts = xyz + (size_t)b * N * 3;
    const float* qq = q + (size_t)bg * 3;
    float qx = qq[0], qy = qq[1], qz = qq[2];
    for (int n = tid; n < N; n += 256) {
        float dx = pts[n * 3] - qx, dy = pts[n * 3 + 1] - qy, dz = pts[n * 3 + 2] - qz;
        dist[n] = dx * dx + dy * dy + dz * dz;
    }
    __syncthreads();
    for (int k = 0; k < KNN_K; k++) {
        float bv = 1e30f; int bi = 0;
        for (int n = tid; n < N; n += 256) {
            float vv = dist[n];
            if (vv < bv) { bv = vv; bi = n; }
        }
        rv[tid] = bv; ri[tid] = bi;
        __syncthreads();
        for (int st = 128; st > 0; st >>= 1) {
            if (tid < st) {
                float ov = rv[tid + st]; int oi = ri[tid + st];
                if (ov < rv[tid] || (ov == rv[tid] && oi < ri[tid])) { rv[tid] = ov; ri[tid] = oi; }
            }
            __syncthreads();
        }
        if (tid == 0) {
            kidx[(size_t)bg * KNN_K + k] = ri[0];
            dist[ri[0]] = 1e30f;
        }
        __syncthreads();
    }
}

// ---------------------------------------------------------------------------
// 6) diff = knn_xyz - lc_xyz; accumulate sum/sumsq for global std (ddof=1)
// ---------------------------------------------------------------------------
__global__ void k_init_stats(float* stats) { stats[0] = 0.f; stats[1] = 0.f; stats[2] = 0.f; }

__global__ void k_diff(const float* __restrict__ xyz, const float* __restrict__ q,
                       const int* __restrict__ kidx, float* __restrict__ dif,
                       float* __restrict__ stats, int N, int G, int total) {
    __shared__ float s1[256], s2[256];
    int tid = threadIdx.x;
    int e = blockIdx.x * blockDim.x + tid;
    float val = 0.f;
    if (e < total) {
        int d = e % 3;
        int t = e / 3;
        int bg = t / KNN_K;
        int b = bg / G;
        int n = kidx[t];
        val = xyz[((size_t)b * N + n) * 3 + d] - q[(size_t)bg * 3 + d];
        dif[e] = val;
    }
    s1[tid] = (e < total) ? val : 0.f;
    s2[tid] = (e < total) ? val * val : 0.f;
    __syncthreads();
    for (int st = 128; st > 0; st >>= 1) {
        if (tid < st) { s1[tid] += s1[tid + st]; s2[tid] += s2[tid + st]; }
        __syncthreads();
    }
    if (tid == 0) {
        atomicAdd(&stats[0], s1[0]);
        atomicAdd(&stats[1], s2[0]);
    }
}

__global__ void k_std(float* stats, float n) {
    float sum = stats[0], sq = stats[1];
    float var = (sq - sum * sum / n) / (n - 1.0f);
    stats[2] = sqrtf(var > 0.f ? var : 0.f);
}

// ---------------------------------------------------------------------------
// 7) gather + concat: kx[t, 0:C] = knn_x ; kx[t, C:2C] = lc_x   (bf16)
// ---------------------------------------------------------------------------
__global__ void k_gather(const uint16_t* __restrict__ act, const int* __restrict__ kidx,
                         const int* __restrict__ fidx, uint16_t* __restrict__ kx,
                         int N, int G, int C, int total) {
    int idx = blockIdx.x * blockDim.x + threadIdx.x;
    if (idx >= total) return;
    int C2 = 2 * C;
    int c = idx % C2;
    int t = idx / C2;
    int bg = t / KNN_K;
    int b = bg / G;
    int n = (c < C) ? kidx[t] : fidx[bg];
    int cc = (c < C) ? c : (c - C);
    kx[(size_t)t * C2 + c] = act[((size_t)b * N + n) * C + cc];
}

// ---------------------------------------------------------------------------
// 8) WMMA GEMM, LDS-blocked:
//    - block = 8 waves, owns one 16-out-channel tile (blockIdx.y) and 512
//      tokens (blockIdx.x). Weight tile (16 x Cin bf16) staged into LDS once
//      per block via TDM tensor_load_to_lds (s_wait_tensorcnt + barrier),
//      fallback: cooperative vector copy.
//    - each wave: 1 A-frag from LDS reused over 4 token tiles (4 accumulators)
//      -> 4x v_wmma_f32_16x16x32_bf16 per 32-deep K step.
//    Epilogue fuses BN scale/shift (+bias), optional residual add, ReLU.
// ---------------------------------------------------------------------------
__global__ __launch_bounds__(256)
void k_gemm_wmma(const uint16_t* __restrict__ X, const uint16_t* __restrict__ W,
                 const float* __restrict__ sc, const float* __restrict__ sh,
                 uint16_t* __restrict__ Y, const uint16_t* __restrict__ R,
                 int T, int Cin, int Cout, int do_relu) {
    __shared__ uint16_t wtile[16 * 1152];   // up to 36KB (stage 3)

    int lane = threadIdx.x & 31;
    int wave = threadIdx.x >> 5;
    int l15 = lane & 15;
    int hi  = lane >> 4;
    int tileM = blockIdx.y;
    const uint16_t* wsrc = W + (size_t)tileM * 16 * Cin;

    // ---- stage 16 x Cin weight tile into LDS ----
#if HAVE_TDM
    if (wave == 0) {                      // one TDM issue per workgroup
        unsigned long long ga = (unsigned long long)(uintptr_t)wsrc;
        unsigned ldsa = (unsigned)(uintptr_t)(void*)wtile;  // LDS aperture: low 32 bits = LDS byte addr
        unsigned ctot = (unsigned)(16 * Cin);               // contiguous 1-D tile, bf16 elements
        v4u g0 = { 1u,                                      // count=1, user descriptor
                   ldsa,
                   (unsigned)ga,
                   ((unsigned)(ga >> 32) & 0x01FFFFFFu) | 0x80000000u };  // addr[56:32] | type=2
        v8i g1 = { (int)0x00010000u,                        // data_size=1 (2 bytes), no multicast
                   (int)((ctot & 0xFFFFu) << 16),           // tensor_dim0[15:0]
                   (int)(((ctot >> 16) & 0xFFFFu) | (1u << 16)), // tensor_dim0[31:16] | tensor_dim1=1
                   (int)((ctot & 0xFFFFu) << 16),           // tile_dim0 = ctot
                   (int)1,                                  // tile_dim1 = 1
                   (int)ctot,                               // tensor_dim0_stride[31:0]
                   0, 0 };
        v4i gz4 = { 0, 0, 0, 0 };
        v8i gz8 = { 0, 0, 0, 0, 0, 0, 0, 0 };
        __builtin_amdgcn_tensor_load_to_lds(g0, g1, gz4, gz4, gz8, 0);
#if __has_builtin(__builtin_amdgcn_s_wait_tensorcnt)
        __builtin_amdgcn_s_wait_tensorcnt(0);
#else
        asm volatile("s_wait_tensorcnt 0x0" ::: "memory");
#endif
    }
#else
    for (int i = threadIdx.x; i < Cin * 4; i += 256)
        ((unsigned long long*)wtile)[i] = ((const unsigned long long*)wsrc)[i];
#endif
    __syncthreads();

    int tok0 = blockIdx.x * 512 + wave * 64;     // 64 tokens per wave (4 tiles)
    const uint16_t* wr = wtile + (size_t)l15 * Cin;
    const uint16_t* xbase = X + (size_t)(tok0 + l15) * Cin;

    v8f acc[4];
#pragma unroll
    for (int j = 0; j < 4; j++) acc[j] = (v8f){0.f, 0.f, 0.f, 0.f, 0.f, 0.f, 0.f, 0.f};

    for (int kk = 0; kk < Cin; kk += 32) {
        Frag16 a;                                 // weights from LDS, reused 4x
        const unsigned long long* pa0 = (const unsigned long long*)(wr + kk + hi * 8);
        const unsigned long long* pa1 = (const unsigned long long*)(wr + kk + 16 + hi * 8);
        a.d[0] = pa0[0]; a.d[1] = pa0[1]; a.d[2] = pa1[0]; a.d[3] = pa1[1];
#pragma unroll
        for (int j = 0; j < 4; j++) {
            const uint16_t* xb = xbase + (size_t)j * 16 * Cin + kk;
            __builtin_prefetch(xb + 64, 0, 0);    // global_prefetch_b8 on streamed tokens
            Frag16 bm;
            const unsigned long long* pb0 = (const unsigned long long*)(xb + hi * 8);
            const unsigned long long* pb1 = (const unsigned long long*)(xb + 16 + hi * 8);
            bm.d[0] = pb0[0]; bm.d[1] = pb0[1]; bm.d[2] = pb1[0]; bm.d[3] = pb1[1];
            acc[j] = __builtin_amdgcn_wmma_f32_16x16x32_bf16(false, a.v, false, bm.v,
                                                             (short)0, acc[j], false, false);
        }
    }

    // ---- epilogue: BN fold, residual, relu, bf16 store ----
    int mbase = tileM * 16 + hi * 8;
    float scv[8], shv[8];
#pragma unroll
    for (int r = 0; r < 8; r++) { scv[r] = sc[mbase + r]; shv[r] = sh[mbase + r]; }
#pragma unroll
    for (int j = 0; j < 4; j++) {
        int t = tok0 + j * 16 + l15;
#pragma unroll
        for (int r = 0; r < 8; r++) {
            int o = mbase + r;
            float v = acc[j][r] * scv[r] + shv[r];
            if (R) v += bf2f(R[(size_t)t * Cout + o]);
            if (do_relu) v = v > 0.f ? v : 0.f;
            Y[(size_t)t * Cout + o] = f2bf(v);
        }
    }
}

// ---------------------------------------------------------------------------
// 9) PosE_Geo:  kw = (lin1_out + pe) * pe
// ---------------------------------------------------------------------------
__global__ void k_pose(const uint16_t* __restrict__ Y, const float* __restrict__ dif,
                       const float* __restrict__ stats, uint16_t* __restrict__ kw,
                       int od, int total) {
    int idx = blockIdx.x * blockDim.x + threadIdx.x;
    if (idx >= total) return;
    int o = idx % od;
    int t = idx / od;
    int f = od / 6;
    int tf = 2 * f;
    int c3 = o / tf;
    int j = o % tf;
    int jj = (j < f) ? j : (j - f);
    float de = powf(1000.0f, (float)jj / (float)f);
    float stdv = stats[2] + 1e-5f;
    float xn = dif[(size_t)t * 3 + c3] / stdv;
    float dv = 100.0f * xn / de;
    float pe = (j < f) ? sinf(dv) : cosf(dv);
    float y = bf2f(Y[idx]);
    kw[idx] = f2bf((y + pe) * pe);
}

// ---------------------------------------------------------------------------
// 10) pool over K:  act[bg,o] = max_k kw + mean_k kw
// ---------------------------------------------------------------------------
__global__ void k_pool(const uint16_t* __restrict__ kw, uint16_t* __restrict__ act,
                       int od, int total) {
    int idx = blockIdx.x * blockDim.x + threadIdx.x;
    if (idx >= total) return;
    int o = idx % od;
    int bg = idx / od;
    size_t base = ((size_t)bg * KNN_K) * od + o;
    float mx = -1e30f, sm = 0.f;
    for (int k = 0; k < KNN_K; k++) {
        float v = bf2f(kw[base + (size_t)k * od]);
        mx = v > mx ? v : mx;
        sm += v;
    }
    act[(size_t)bg * od + o] = f2bf(mx + sm * (1.0f / KNN_K));
}

// ---------------------------------------------------------------------------
// 11) final pool over G -> d_out (f32, B x 1152)
// ---------------------------------------------------------------------------
__global__ void k_final(const uint16_t* __restrict__ act, float* __restrict__ out,
                        int G, int od, int total) {
    int idx = blockIdx.x * blockDim.x + threadIdx.x;
    if (idx >= total) return;
    int o = idx % od;
    int b = idx / od;
    size_t base = ((size_t)b * G) * od + o;
    float mx = -1e30f, sm = 0.f;
    for (int g = 0; g < G; g++) {
        float v = bf2f(act[base + (size_t)g * od]);
        mx = v > mx ? v : mx;
        sm += v;
    }
    out[idx] = mx + sm / (float)G;
}

// ---------------------------------------------------------------------------
// Host orchestration
// ---------------------------------------------------------------------------
extern "C" void kernel_launch(void* const* d_in, const int* in_sizes, int n_in,
                              void* d_out, int out_size, void* d_ws, size_t ws_size,
                              hipStream_t stream) {
    (void)in_sizes; (void)n_in; (void)out_size; (void)ws_size;

    const int OD[4] = {144, 288, 576, 1152};
    const int GR[4] = {1024, 512, 256, 128};
    const int B = BATCH;

    auto P = [&](int i) -> const float* { return (const float*)d_in[i]; };
    const float* xyz0 = P(0);
    const float* x0   = P(1);
    const float* ew   = P(2);

    char* ws = (char*)d_ws;
    size_t off = 0;
    auto alloc = [&](size_t bytes) -> void* {
        off = (off + 255) & ~(size_t)255;
        void* p = (void*)(ws + off);
        off += bytes;
        return p;
    };

    const size_t ACT_ELEMS = (size_t)B * NPTS * 72;
    const size_t BIG_ELEMS = (size_t)B * 1024 * KNN_K * 144;

    uint16_t* actA = (uint16_t*)alloc(ACT_ELEMS * 2);
    uint16_t* actB = (uint16_t*)alloc(ACT_ELEMS * 2);
    float* xyzA = (float*)alloc((size_t)B * NPTS * 3 * 4);
    float* xyzB = (float*)alloc((size_t)B * NPTS * 3 * 4);
    int* fidx = (int*)alloc((size_t)B * 1024 * 4);
    int* kidx = (int*)alloc((size_t)B * 1024 * KNN_K * 4);
    float* dif = (float*)alloc((size_t)B * 1024 * KNN_K * 3 * 4);
    float* stats = (float*)alloc(256);
    uint16_t* kx  = (uint16_t*)alloc(BIG_ELEMS * 2);
    uint16_t* ybf = (uint16_t*)alloc(BIG_ELEMS * 2);

    uint16_t* lin1bf[4]; uint16_t* w1bf[4]; uint16_t* w2bf[4];
    float *sL[4], *tL[4], *s1[4], *t1[4], *s2[4], *t2[4];
    float* sE = (float*)alloc(72 * 4);
    float* tE = (float*)alloc(72 * 4);
    for (int i = 0; i < 4; i++) {
        int od = OD[i];
        lin1bf[i] = (uint16_t*)alloc((size_t)od * od * 2);
        w1bf[i]   = (uint16_t*)alloc((size_t)od * od * 2);
        w2bf[i]   = (uint16_t*)alloc((size_t)od * od * 2);
        sL[i] = (float*)alloc(od * 4); tL[i] = (float*)alloc(od * 4);
        s1[i] = (float*)alloc(od * 4); t1[i] = (float*)alloc(od * 4);
        s2[i] = (float*)alloc(od * 4); t2[i] = (float*)alloc(od * 4);
    }

    // ---- prep: weight conversion + BN folding ----
    k_prep_ss<<<1, 128, 0, stream>>>(P(3), P(4), P(5), P(6), nullptr, sE, tE, 72);
    for (int i = 0; i < 4; i++) {
        int od = OD[i];
        int base = 7 + i * 17;
        int nw = od * od;
        int gcv = (nw + 255) / 256;
        k_cvt_bf16<<<gcv, 256, 0, stream>>>(P(base + 0),  lin1bf[i], nw);
        k_cvt_bf16<<<gcv, 256, 0, stream>>>(P(base + 5),  w1bf[i],   nw);
        k_cvt_bf16<<<gcv, 256, 0, stream>>>(P(base + 11), w2bf[i],   nw);
        int gss = (od + 255) / 256;
        k_prep_ss<<<gss, 256, 0, stream>>>(P(base + 1), P(base + 2), P(base + 3), P(base + 4),
                                           nullptr, sL[i], tL[i], od);
        k_prep_ss<<<gss, 256, 0, stream>>>(P(base + 7), P(base + 8), P(base + 9), P(base + 10),
                                           P(base + 6), s1[i], t1[i], od);
        k_prep_ss<<<gss, 256, 0, stream>>>(P(base + 13), P(base + 14), P(base + 15), P(base + 16),
                                           P(base + 12), s2[i], t2[i], od);
    }

    // ---- embed ----
    {
        int total = B * NPTS * 72;
        k_embed<<<(total + 255) / 256, 256, 0, stream>>>(x0, ew, sE, tE, actA, NPTS);
    }

    // ---- stages ----
    const float* xyz_cur = xyz0;
    for (int i = 0; i < 4; i++) {
        int Ncur = (i == 0) ? NPTS : GR[i - 1];
        int G = GR[i];
        int C = (i == 0) ? 72 : OD[i - 1];
        int od = OD[i];                 // == 2*C
        int T = B * G * KNN_K;
        float* xyz_next = (i % 2 == 0) ? xyzA : xyzB;
        const uint16_t* act_cur = (i % 2 == 0) ? actA : actB;
        uint16_t* act_next = (i % 2 == 0) ? actB : actA;

        k_fps<<<B, 256, 0, stream>>>(xyz_cur, Ncur, G, fidx, xyz_next);
        k_knn<<<B * G, 256, 0, stream>>>(xyz_cur, xyz_next, Ncur, G, kidx);

        k_init_stats<<<1, 1, 0, stream>>>(stats);
        int td = T * 3;
        k_diff<<<(td + 255) / 256, 256, 0, stream>>>(xyz_cur, xyz_next, kidx, dif, stats,
                                                     Ncur, G, td);
        k_std<<<1, 1, 0, stream>>>(stats, (float)td);

        int tg = T * 2 * C;
        k_gather<<<(tg + 255) / 256, 256, 0, stream>>>(act_cur, kidx, fidx, kx, Ncur, G, C, tg);

        dim3 gg(T / 512, od / 16);
        // lin1: relu(bn(W.kx))
        k_gemm_wmma<<<gg, 256, 0, stream>>>(kx, lin1bf[i], sL[i], tL[i], ybf, nullptr,
                                            T, od, od, 1);
        // PosE modulation -> kw (reuse kx buffer)
        int tp = T * od;
        k_pose<<<(tp + 255) / 256, 256, 0, stream>>>(ybf, dif, stats, kx, od, tp);
        // block: h = relu(bn1(W1.kw + b1))
        k_gemm_wmma<<<gg, 256, 0, stream>>>(kx, w1bf[i], s1[i], t1[i], ybf, nullptr,
                                            T, od, od, 1);
        // kw = relu(bn2(W2.h + b2) + kw)   (in-place residual)
        k_gemm_wmma<<<gg, 256, 0, stream>>>(ybf, w2bf[i], s2[i], t2[i], kx, kx,
                                            T, od, od, 1);
        // pool over K
        int pt = B * G * od;
        k_pool<<<(pt + 255) / 256, 256, 0, stream>>>(kx, act_next, od, pt);

        xyz_cur = xyz_next;
    }

    // ---- final pool over G=128 -> (B, 1152) f32 ----
    {
        int total = B * 1152;
        k_final<<<(total + 255) / 256, 256, 0, stream>>>(actA, (float*)d_out, 128, 1152, total);
    }
}
